// Speller_44865228374357
// MI455X (gfx1250) — matrix operand order repaired and checked by reading, not compiled, not run on hardware
//
#include <hip/hip_runtime.h>
#include <math.h>

// Problem constants (from reference)
#define B_   64
#define T_   256
#define ET_  1024
#define U_   512
#define V_   46
#define N1_  2048      // 4*U gate width
#define K1_  1568      // padded K for LSTM1: [h1(512), y(1), ctx(512), h1(512), pad(31)]
#define K1T_ 49        // K1_/32
#define K2_  1024      // LSTM2: [ctx(512), h2(512)]
#define K2T_ 32

typedef __attribute__((ext_vector_type(16))) __bf16 v16bf;
typedef __attribute__((ext_vector_type(8)))  __bf16 v8bf;
typedef __attribute__((ext_vector_type(8)))  float  v8f;

__device__ __forceinline__ float sigmoidf_(float x) { return 1.f / (1.f + __expf(-x)); }

// ---------------------------------------------------------------------------
// GEMM: C(MxN,f32) = A(MxK,bf16 row-major) * Bpacked(bf16, WMMA-tile layout)
// One wave per 16x16 C tile. K = Kt*32, WMMA 16x16x32 bf16 -> f32.
// Bpacked layout: [kt][nt][lane][16 bf16] in contiguous k order, same as A:
//   lane: column n = lane&15 ; K half = (lane&16)?16:0 ; element e -> K = half+e
// A per-lane: row m = tileM*16 + (lane&15); koff = (lane&16)?16:0;
//   elements 0..15 -> K = kt*32+koff+e
// ---------------------------------------------------------------------------
__global__ void gemm_bf16_wmma(const __bf16* __restrict__ A,
                               const __bf16* __restrict__ Bp,
                               float* __restrict__ C,
                               int N, int Kt)
{
    const int lane = threadIdx.x & 31;
    const int wave = threadIdx.x >> 5;
    const int tile = blockIdx.x * (blockDim.x >> 5) + wave;
    const int NT   = N >> 4;
    const int mt   = tile / NT;
    const int nt   = tile - mt * NT;
    const int lda  = Kt << 5;
    const int m    = (mt << 4) + (lane & 15);
    const int koff = (lane & 16) ? 16 : 0;

    const __bf16* arow = A + (size_t)m * lda + koff;
    const __bf16* bptr = Bp + (((size_t)nt * 32 + lane) << 4);
    const size_t  bstride = (size_t)NT * 32 * 16;

    v8f acc = {};
    for (int kt = 0; kt < Kt; ++kt) {
        v8bf lo = *(const v8bf*)(arow + (kt << 5));
        v8bf hi = *(const v8bf*)(arow + (kt << 5) + 8);
        v16bf a = __builtin_shufflevector(lo, hi, 0,1,2,3,4,5,6,7,8,9,10,11,12,13,14,15);
        v16bf b = *(const v16bf*)(bptr + (size_t)kt * bstride);
        acc = __builtin_amdgcn_wmma_f32_16x16x32_bf16(false, a, false, b,
                                                      (short)0, acc, false, false);
    }
    // C/D layout: VGPR j -> row mt*16 + j + ((lane&16)?8:0); col nt*16 + (lane&15)
    const int row0 = (mt << 4) + ((lane & 16) ? 8 : 0);
    const int col  = (nt << 4) + (lane & 15);
    float* cp = C + (size_t)row0 * N + col;
#pragma unroll
    for (int j = 0; j < 8; ++j) cp[(size_t)j * N] = acc[j];
}

// ---------------------------------------------------------------------------
// Pack fp32 weights [Wtop; Wbot; zero-pad] (K x N) into bf16 WMMA B-tile layout.
// One thread per (kt, nt, lane) writes its 16 contiguous bf16 (32 bytes).
// ---------------------------------------------------------------------------
__global__ void pack_b_tiles(const float* __restrict__ Wtop, int rows_top,
                             const float* __restrict__ Wbot, int rows_bot,
                             __bf16* __restrict__ Bp, int N, int Kt)
{
    int gid = blockIdx.x * blockDim.x + threadIdx.x;
    int NT = N >> 4;
    int total = Kt * NT * 32;
    if (gid >= total) return;
    int lane = gid & 31;
    int tile = gid >> 5;
    int nt = tile % NT;
    int kt = tile / NT;
    int n = (nt << 4) + (lane & 15);
    int kbase = (kt << 5) + ((lane & 16) ? 16 : 0);
    __bf16 vals[16];
#pragma unroll
    for (int e = 0; e < 16; ++e) {
        int k = kbase + e;
        float f = 0.f;
        if (k < rows_top)                 f = Wtop[(size_t)k * N + n];
        else if (k < rows_top + rows_bot) f = Wbot[(size_t)(k - rows_top) * N + n];
        vals[e] = (__bf16)f;
    }
    __bf16* dst = Bp + ((size_t)gid << 4);
#pragma unroll
    for (int e = 0; e < 16; ++e) dst[e] = vals[e];
}

// ---------------------------------------------------------------------------
// Prep: zero states h1,c1,h2,c2,ctx; zero x2; init x1 = [0, y_0, 0, 0, pad0].
// ---------------------------------------------------------------------------
__global__ void prep_init(__bf16* __restrict__ x1, __bf16* __restrict__ x2,
                          float* __restrict__ st, const float* __restrict__ y)
{
    int gid = blockIdx.x * blockDim.x + threadIdx.x;
    if (gid < B_ * K1_) {
        int col = gid % K1_;
        int b   = gid / K1_;
        x1[gid] = (col == 512) ? (__bf16)y[(size_t)b * T_] : (__bf16)0.f;
    }
    if (gid < B_ * K2_) x2[gid] = (__bf16)0.f;
    if (gid < 5 * B_ * U_) st[gid] = 0.f;
}

// ---------------------------------------------------------------------------
// LSTM1 gates (Keras order i,f,g,o) + repack x1 slots for next step.
// ---------------------------------------------------------------------------
__global__ void lstm_gates1(const float* __restrict__ z, const float* __restrict__ bias,
                            float* __restrict__ c, float* __restrict__ h,
                            __bf16* __restrict__ x1, const float* __restrict__ y, int tnext)
{
    int gid = blockIdx.x * blockDim.x + threadIdx.x;
    if (gid >= B_ * U_) return;
    int b = gid / U_;
    int n = gid - b * U_;
    const float* zb = z + (size_t)b * N1_;
    float zi = zb[n]           + bias[n];
    float zf = zb[n +   U_]    + bias[n +   U_];
    float zg = zb[n + 2 * U_]  + bias[n + 2 * U_];
    float zo = zb[n + 3 * U_]  + bias[n + 3 * U_];
    float cn = sigmoidf_(zf) * c[gid] + sigmoidf_(zi) * tanhf(zg);
    float hn = sigmoidf_(zo) * tanhf(cn);
    c[gid] = cn;
    h[gid] = hn;
    __bf16  hb = (__bf16)hn;
    __bf16* xb = x1 + (size_t)b * K1_;
    xb[n]        = hb;   // W1 h1 slot
    xb[1025 + n] = hb;   // U1 h1 slot
    if (n == 0) {
        float yv = (tnext < T_) ? y[(size_t)b * T_ + tnext] : 0.f;
        xb[512] = (__bf16)yv;
    }
}

// ---------------------------------------------------------------------------
// LSTM2 gates + repack h2 slot of x2 for next step.
// ---------------------------------------------------------------------------
__global__ void lstm_gates2(const float* __restrict__ z, const float* __restrict__ bias,
                            float* __restrict__ c, float* __restrict__ h,
                            __bf16* __restrict__ x2)
{
    int gid = blockIdx.x * blockDim.x + threadIdx.x;
    if (gid >= B_ * U_) return;
    int b = gid / U_;
    int n = gid - b * U_;
    const float* zb = z + (size_t)b * N1_;
    float zi = zb[n]           + bias[n];
    float zf = zb[n +   U_]    + bias[n +   U_];
    float zg = zb[n + 2 * U_]  + bias[n + 2 * U_];
    float zo = zb[n + 3 * U_]  + bias[n + 3 * U_];
    float cn = sigmoidf_(zf) * c[gid] + sigmoidf_(zi) * tanhf(zg);
    float hn = sigmoidf_(zo) * tanhf(cn);
    c[gid] = cn;
    h[gid] = hn;
    x2[(size_t)b * K2_ + 512 + n] = (__bf16)hn;
}

// ---------------------------------------------------------------------------
// Fused attention: scores = h1·h  ->  softmax  ->  ctx = attn·h.
// One block (256 threads) per batch element. h stays fp32 (L2-resident).
// Also writes bf16 ctx slots into x1 (next step) and x2 (this step).
// ---------------------------------------------------------------------------
__global__ void attn_softmax_ctx(const float* __restrict__ h,
                                 const float* __restrict__ h1,
                                 float* __restrict__ ctx,
                                 __bf16* __restrict__ x1,
                                 __bf16* __restrict__ x2)
{
    __shared__ float qs[U_];
    __shared__ float sc[ET_];
    __shared__ float red[8];
    __shared__ float bc;
    const int tid  = threadIdx.x;
    const int lane = tid & 31;
    const int wid  = tid >> 5;
    const int b    = blockIdx.x;
    const float* hb = h  + (size_t)b * ET_ * U_;
    const float* q  = h1 + (size_t)b * U_;

    for (int i = tid; i < U_; i += 256) qs[i] = q[i];
    __syncthreads();

    // scores
    for (int t = tid; t < ET_; t += 256) {
        const float* row = hb + (size_t)t * U_;
        float acc = 0.f;
#pragma unroll 4
        for (int d = 0; d < U_; d += 4) {
            float4 r = *(const float4*)(row + d);
            acc += r.x * qs[d] + r.y * qs[d + 1] + r.z * qs[d + 2] + r.w * qs[d + 3];
        }
        sc[t] = acc;
    }
    __syncthreads();

    // softmax: block max
    float m = -3.4e38f;
    for (int t = tid; t < ET_; t += 256) m = fmaxf(m, sc[t]);
    for (int o = 16; o > 0; o >>= 1) m = fmaxf(m, __shfl_xor(m, o, 32));
    if (lane == 0) red[wid] = m;
    __syncthreads();
    if (tid == 0) {
        float mm = red[0];
        for (int i = 1; i < 8; ++i) mm = fmaxf(mm, red[i]);
        bc = mm;
    }
    __syncthreads();
    m = bc;

    // softmax: exp + block sum
    float s = 0.f;
    for (int t = tid; t < ET_; t += 256) {
        float e = __expf(sc[t] - m);
        sc[t] = e;
        s += e;
    }
    for (int o = 16; o > 0; o >>= 1) s += __shfl_xor(s, o, 32);
    if (lane == 0) red[wid] = s;
    __syncthreads();
    if (tid == 0) {
        float ss = 0.f;
        for (int i = 0; i < 8; ++i) ss += red[i];
        bc = ss;
    }
    __syncthreads();
    const float inv = 1.f / bc;

    // ctx: each thread owns columns tid and tid+256
    float a0 = 0.f, a1 = 0.f;
    for (int t = 0; t < ET_; ++t) {
        const float* row = hb + (size_t)t * U_;
        if (t + 8 < ET_) __builtin_prefetch(row + 8 * U_, 0, 0);
        float w = sc[t] * inv;
        a0 += w * row[tid];
        a1 += w * row[tid + 256];
    }
    float* cb = ctx + (size_t)b * U_;
    cb[tid]       = a0;
    cb[tid + 256] = a1;
    __bf16* x1b = x1 + (size_t)b * K1_ + 513;   // ctx slot in x1
    x1b[tid]       = (__bf16)a0;
    x1b[tid + 256] = (__bf16)a1;
    __bf16* x2b = x2 + (size_t)b * K2_;         // ctx slot in x2
    x2b[tid]       = (__bf16)a0;
    x2b[tid + 256] = (__bf16)a1;
}

// ---------------------------------------------------------------------------
// logits = [h2, ctx] @ Wd + bd ; softmax over V=46 ; write out[b][t][:].
// One wave per batch element (lane owns columns lane and lane+32).
// ---------------------------------------------------------------------------
__global__ void logits_softmax(const float* __restrict__ h2, const float* __restrict__ ctx,
                               const float* __restrict__ Wd, const float* __restrict__ bd,
                               float* __restrict__ out, int t)
{
    int gt   = blockIdx.x * blockDim.x + threadIdx.x;
    int wave = gt >> 5;
    int lane = gt & 31;
    if (wave >= B_) return;
    int b = wave;
    const float* hb = h2  + (size_t)b * U_;
    const float* cb = ctx + (size_t)b * U_;
    int  v0   = lane;
    int  v1   = lane + 32;
    bool val1 = (v1 < V_);
    float a0 = bd[v0];
    float a1 = val1 ? bd[v1] : 0.f;
    for (int k = 0; k < U_; ++k) {
        float xv = hb[k];
        const float* wr = Wd + (size_t)k * V_;
        a0 += xv * wr[v0];
        if (val1) a1 += xv * wr[v1];
    }
    for (int k = 0; k < U_; ++k) {
        float xv = cb[k];
        const float* wr = Wd + (size_t)(U_ + k) * V_;
        a0 += xv * wr[v0];
        if (val1) a1 += xv * wr[v1];
    }
    float m = fmaxf(a0, val1 ? a1 : -3.4e38f);
    for (int o = 16; o > 0; o >>= 1) m = fmaxf(m, __shfl_xor(m, o, 32));
    float e0 = __expf(a0 - m);
    float e1 = val1 ? __expf(a1 - m) : 0.f;
    float s = e0 + e1;
    for (int o = 16; o > 0; o >>= 1) s += __shfl_xor(s, o, 32);
    float inv = 1.f / s;
    float* ob = out + ((size_t)b * T_ + t) * V_;
    ob[v0] = e0 * inv;
    if (val1) ob[v1] = e1 * inv;
}

// ---------------------------------------------------------------------------
extern "C" void kernel_launch(void* const* d_in, const int* in_sizes, int n_in,
                              void* d_out, int out_size, void* d_ws, size_t ws_size,
                              hipStream_t stream)
{
    (void)in_sizes; (void)n_in; (void)out_size; (void)ws_size;

    const float* h  = (const float*)d_in[0];
    const float* y  = (const float*)d_in[1];
    const float* W1 = (const float*)d_in[2];
    const float* U1 = (const float*)d_in[3];
    const float* b1 = (const float*)d_in[4];
    const float* W2 = (const float*)d_in[5];
    const float* U2 = (const float*)d_in[6];
    const float* b2 = (const float*)d_in[7];
    const float* Wd = (const float*)d_in[8];
    const float* bd = (const float*)d_in[9];
    float* out = (float*)d_out;

    // Workspace carve-up (~12.5 MB, all offsets 256B aligned)
    char* p = (char*)d_ws;
    __bf16* Wc1p = (__bf16*)p; p += (size_t)K1T_ * (N1_ >> 4) * 32 * 16 * 2; // 6.4 MB
    __bf16* Wc2p = (__bf16*)p; p += (size_t)K2T_ * (N1_ >> 4) * 32 * 16 * 2; // 4.0 MB
    __bf16* x1   = (__bf16*)p; p += (size_t)B_ * K1_ * 2;
    __bf16* x2   = (__bf16*)p; p += (size_t)B_ * K2_ * 2;
    float*  z1   = (float*)p;  p += (size_t)B_ * N1_ * 4;
    float*  z2   = (float*)p;  p += (size_t)B_ * N1_ * 4;
    float*  st   = (float*)p;  // 5 * B*U states
    float* h1 = st;
    float* c1 = st + 1 * B_ * U_;
    float* h2 = st + 2 * B_ * U_;
    float* c2 = st + 3 * B_ * U_;
    float* cx = st + 4 * B_ * U_;

    // --- prep: pack weights to bf16 WMMA tile layout, init states/activations
    {
        int tot1 = K1T_ * (N1_ >> 4) * 32;
        pack_b_tiles<<<(tot1 + 255) / 256, 256, 0, stream>>>(W1, 1025, U1, 512, Wc1p, N1_, K1T_);
        int tot2 = K2T_ * (N1_ >> 4) * 32;
        pack_b_tiles<<<(tot2 + 255) / 256, 256, 0, stream>>>(W2, 512, U2, 512, Wc2p, N1_, K2T_);
        prep_init<<<(5 * B_ * U_ + 255) / 256, 256, 0, stream>>>(x1, x2, st, y);
    }

    // (M/16)*(N/16) = 4*128 = 512 tiles; 8 waves/block -> exact grid of 64 blocks
    const int gemm_blocks  = ((B_ / 16) * (N1_ >> 4)) / 8;
    const int gates_blocks = (B_ * U_ + 255) / 256;

    for (int t = 0; t < T_; ++t) {
        gemm_bf16_wmma<<<gemm_blocks, 256, 0, stream>>>(x1, Wc1p, z1, N1_, K1T_);
        lstm_gates1<<<gates_blocks, 256, 0, stream>>>(z1, b1, c1, h1, x1, y, t + 1);
        attn_softmax_ctx<<<B_, 256, 0, stream>>>(h, h1, cx, x1, x2);
        gemm_bf16_wmma<<<gemm_blocks, 256, 0, stream>>>(x2, Wc2p, z2, N1_, K2T_);
        lstm_gates2<<<gates_blocks, 256, 0, stream>>>(z2, b2, c2, h2, x2);
        logits_softmax<<<(B_ * 32 + 255) / 256, 256, 0, stream>>>(h2, cx, Wd, bd, out, t);
    }
}
